// Mamba2Block_83631603188089
// MI455X (gfx1250) — matrix operand order repaired and checked by reading, not compiled, not run on hardware
//
#include <hip/hip_runtime.h>
#include <hip/hip_bf16.h>

// ---------------- problem constants (from reference) ----------------
#define B_  8
#define L_  4096
#define D_  768
#define N_  64
#define K_  4
#define P_  64
#define DI_ 1536            // E*D
#define H_  24              // DI/P
#define CD_ 1664            // DI + 2N
#define DIP_ 3224           // 2*DI + 2N + H
#define MROWS_ (B_ * L_)    // 32768

// ---------------- WMMA vector types ----------------
typedef __attribute__((ext_vector_type(16))) __bf16         v16bf;
typedef __attribute__((ext_vector_type(8)))  float          v8f;
typedef __attribute__((ext_vector_type(8)))  unsigned short v8us;

union ABfrag { v16bf b; v8us h[2]; };

#define AS3 __attribute__((address_space(3)))

// LDS byte offset of a shared-memory pointer
static __device__ __forceinline__ unsigned lds_addr32(const void* p) {
    return (unsigned)(unsigned long long)(AS3 const void*)p;
}

// CDNA5 async DMA: 16B global -> LDS, tracked by ASYNCcnt (no VGPR transit)
static __device__ __forceinline__ void async_ld16(unsigned ldsOff, const void* g) {
    asm volatile("global_load_async_to_lds_b128 %0, %1, off"
                 :: "v"(ldsOff), "v"((unsigned long long)g)
                 : "memory");
}

static __device__ __forceinline__ void wait_async0() {
#if defined(__has_builtin) && __has_builtin(__builtin_amdgcn_s_wait_asynccnt)
    __builtin_amdgcn_s_wait_asynccnt(0);
#else
    asm volatile("s_wait_asynccnt 0x0" ::: "memory");
#endif
}

// fp32 -> bf16 round-to-nearest-even
static __device__ __forceinline__ unsigned short f2bf(float f) {
    unsigned int u = __builtin_bit_cast(unsigned int, f);
    u += 0x7FFFu + ((u >> 16) & 1u);
    return (unsigned short)(u >> 16);
}

static __device__ __forceinline__ float siluf(float v) {
    return v / (1.0f + __expf(-v));
}

// ---------------- 1) generic fp32 -> bf16 cast ----------------
__global__ void k_cast_bf16(const float* __restrict__ src,
                            unsigned short* __restrict__ dst, long long n) {
    long long i = (long long)blockIdx.x * 256 + threadIdx.x;
    if (i < n) dst[i] = f2bf(src[i]);
}

// ---------------- 2) LayerNorm + cast to bf16 ----------------
__global__ __launch_bounds__(256) void k_ln_cast(
    const float* __restrict__ x, const float* __restrict__ w,
    const float* __restrict__ bias, unsigned short* __restrict__ out) {
    int row = blockIdx.x, t = threadIdx.x;
    const float* xr = x + (size_t)row * D_;
    float s = 0.f, s2 = 0.f;
    for (int d = t; d < D_; d += 256) { float v = xr[d]; s += v; s2 += v * v; }
    __shared__ float r1[256], r2[256];
    r1[t] = s; r2[t] = s2; __syncthreads();
    for (int o = 128; o > 0; o >>= 1) {
        if (t < o) { r1[t] += r1[t + o]; r2[t] += r2[t + o]; }
        __syncthreads();
    }
    float mu  = r1[0] / (float)D_;
    float var = r2[0] / (float)D_ - mu * mu;
    float inv = rsqrtf(var + 1e-5f);
    for (int d = t; d < D_; d += 256) {
        float v = (xr[d] - mu) * inv * w[d] + bias[d];
        out[(size_t)row * D_ + d] = f2bf(v);
    }
}

// ---------------- 3) WMMA bf16 GEMM: C = A(MxK) * W(NcxK)^T [+res] ----------------
// 128x64 macro-tile, BK=32, 8 waves, each wave computes a 16x64 strip
// (4 accumulators, A fragment reused). Double-buffered LDS filled by
// GLOBAL_LOAD_ASYNC_TO_LDS_B128 (ASYNCcnt) so DMA of tile k+1 overlaps WMMA
// of tile k.
#define BM 128
#define BN 64
#define BK 32
#define LDT 40   // padded LDS row (ushorts): 80B rows keep 16B-aligned segments

__global__ __launch_bounds__(256) void k_gemm_bf16(
    const unsigned short* __restrict__ A,   // M x K bf16 row-major
    const unsigned short* __restrict__ W,   // Nc x K bf16 row-major
    float* __restrict__ C,                  // M x Nc
    const float* __restrict__ res,          // residual (M x Nc) or unused
    int M, int K, int Nc, int hasRes) {
    __shared__ __align__(16) unsigned short As[2][BM][LDT];
    __shared__ __align__(16) unsigned short Bs[2][BN][LDT];

    const int t    = threadIdx.x;
    const int wave = t >> 5, lane = t & 31;
    const int m0   = blockIdx.y * BM;
    const int n0   = blockIdx.x * BN;
    const int hi   = lane >> 4;      // lane half
    const int l16  = lane & 15;

    // cooperative-load task: row ra (0..63), 8-ushort segment sa
    const int ra = t >> 2;
    const int sa = (t & 3) * 8;
    const bool bok = (n0 + ra) < Nc;

    // Nc tail rows of the B tile: zero once; those lanes never async-load,
    // so the zeros persist across all K-steps and both buffers.
    if (!bok) {
        v8us z = {};
        *(v8us*)&Bs[0][ra][sa] = z;
        *(v8us*)&Bs[1][ra][sa] = z;
    }

    auto stage = [&](int buf, int k0) {
        async_ld16(lds_addr32(&As[buf][ra][sa]),
                   A + (size_t)(m0 + ra) * K + k0 + sa);
        async_ld16(lds_addr32(&As[buf][ra + 64][sa]),
                   A + (size_t)(m0 + ra + 64) * K + k0 + sa);
        if (bok)
            async_ld16(lds_addr32(&Bs[buf][ra][sa]),
                       W + (size_t)(n0 + ra) * K + k0 + sa);
    };

    v8f acc[4] = {{}, {}, {}, {}};

    stage(0, 0);
    wait_async0();
    __syncthreads();

    int buf = 0;
    for (int k0 = 0; k0 < K; k0 += BK) {
        if (k0 + BK < K) stage(buf ^ 1, k0 + BK);          // overlap DMA
        if (k0 + 2 * BK < K)                               // global_prefetch_b8
            __builtin_prefetch(A + (size_t)(m0 + ra) * K + k0 + 2 * BK + sa, 0, 1);

        // A fragment (16x32, ISA lane-half K swizzle), reused across 4 B tiles
        ABfrag a;
        const int am = wave * 16 + l16;
        a.h[0] = *(const v8us*)&As[buf][am][8 * hi];        // K 0..7   (+8*hi)
        a.h[1] = *(const v8us*)&As[buf][am][16 + 8 * hi];   // K 16..23 (+8*hi)
        #pragma unroll
        for (int nt = 0; nt < 4; ++nt) {
            ABfrag bf;
            const int nr = nt * 16 + l16;
            bf.h[0] = *(const v8us*)&Bs[buf][nr][8 * hi];
            bf.h[1] = *(const v8us*)&Bs[buf][nr][16 + 8 * hi];
            acc[nt] = __builtin_amdgcn_wmma_f32_16x16x32_bf16(
                false, a.b, false, bf.b, (short)0, acc[nt], false, false);
        }

        wait_async0();       // our async loads for buf^1 done
        __syncthreads();     // everyone's done; safe to read buf^1 / refill buf
        buf ^= 1;
    }

    // store: f32 D layout: VGPR rr -> row rr (+8 for lane half), col = lane%16
    #pragma unroll
    for (int nt = 0; nt < 4; ++nt) {
        int n = n0 + nt * 16 + l16;
        if (n < Nc) {
            #pragma unroll
            for (int rr = 0; rr < 8; ++rr) {
                int m = m0 + wave * 16 + rr + 8 * hi;
                size_t idx = (size_t)m * Nc + n;
                float v = acc[nt][rr];
                if (hasRes) v += res[idx];
                C[idx] = v;
            }
        }
    }
}

// ---------------- 4) depthwise causal conv (K=4) + bias + SiLU ----------------
__global__ void k_conv_silu(const float* __restrict__ Z,  // M x DIP
                            const float* __restrict__ cw, // CD x K
                            const float* __restrict__ cb, // CD
                            float* __restrict__ out) {    // M x CD
    long long idx = (long long)blockIdx.x * 256 + threadIdx.x;
    if (idx >= (long long)MROWS_ * CD_) return;
    int c = (int)(idx % CD_);
    long long row = idx / CD_;
    int l = (int)(row % L_);
    float acc = cb[c];
    #pragma unroll
    for (int j = 0; j < K_; ++j) {
        int dl = l - (K_ - 1) + j;
        if (dl >= 0)
            acc += cw[c * K_ + j] * Z[(row - (K_ - 1) + j) * (long long)DIP_ + DI_ + c];
    }
    out[row * (long long)CD_ + c] = siluf(acc);
}

// ---------------- 5) dt = softplus(.), dA = exp(-exp(A_log)*dt) ----------------
__global__ void k_dt(const float* __restrict__ Z, const float* __restrict__ dt_bias,
                     const float* __restrict__ A_log,
                     float* __restrict__ dtb, float* __restrict__ dAb) {
    int idx = blockIdx.x * 256 + threadIdx.x;
    if (idx >= MROWS_ * H_) return;
    int h = idx % H_;
    long long row = idx / H_;
    float v  = Z[row * (long long)DIP_ + DI_ + CD_ + h] + dt_bias[h];
    float sp = (v > 20.f) ? v : log1pf(__expf(v));
    dtb[idx] = sp;
    dAb[idx] = __expf(-__expf(A_log[h]) * sp);
}

// ---------------- 6) selective scan: one block per (b,h), state in registers ----
__global__ __launch_bounds__(256) void k_scan(
    const float* __restrict__ xBC,  // M x CD  (silu'd conv out)
    const float* __restrict__ dtb,  // M x H
    const float* __restrict__ dAb,  // M x H
    const float* __restrict__ Dsk,  // H
    float* __restrict__ Y) {        // M x DI
    const int b = blockIdx.x / H_;
    const int h = blockIdx.x % H_;
    const int t = threadIdx.x;
    const int p = t & 63;          // state row
    const int ng = t >> 6;         // n-group (4 groups of 16)
    const int nb = ng * 16;

    __shared__ float sx[P_], sB[N_], sC[N_], sdd[2];
    __shared__ float ypart[4][P_];

    float hst[16];
    #pragma unroll
    for (int j = 0; j < 16; ++j) hst[j] = 0.f;
    const float dskip = Dsk[h];

    for (int l = 0; l < L_; ++l) {
        long long row = (long long)b * L_ + l;
        const float* xr = xBC + row * CD_;
        if (t < 64)       sx[t]       = xr[h * P_ + t];
        else if (t < 128) sB[t - 64]  = xr[DI_ + (t - 64)];
        else if (t < 192) sC[t - 128] = xr[DI_ + N_ + (t - 128)];
        else if (t == 192) { sdd[0] = dAb[row * H_ + h]; sdd[1] = dtb[row * H_ + h]; }
        __syncthreads();

        float dA = sdd[0], dtv = sdd[1];
        float xp = sx[p];
        float coef = dtv * xp;
        float acc = 0.f;
        #pragma unroll
        for (int j = 0; j < 16; ++j) {
            float hn = hst[j] * dA + coef * sB[nb + j];
            hst[j] = hn;
            acc += hn * sC[nb + j];
        }
        ypart[ng][p] = acc;
        __syncthreads();

        if (t < 64) {
            float y = ypart[0][t] + ypart[1][t] + ypart[2][t] + ypart[3][t]
                    + dskip * xp;  // p == t here, xp is register copy of sx[t]
            Y[row * DI_ + h * P_ + t] = y;
        }
    }
}

// ---------------- 7) gating + RMSNorm + cast to bf16 ----------------
__global__ __launch_bounds__(256) void k_gate(
    const float* __restrict__ Y,  // M x DI
    const float* __restrict__ Z,  // M x DIP (z = cols 0..DI-1)
    const float* __restrict__ nw, // DI
    unsigned short* __restrict__ G) { // M x DI bf16
    int row = blockIdx.x, t = threadIdx.x;
    const float* yr = Y + (size_t)row * DI_;
    const float* zr = Z + (size_t)row * DIP_;
    float gv[6]; float s2 = 0.f;
    #pragma unroll
    for (int i = 0; i < 6; ++i) {
        int e = t + i * 256;
        float g = yr[e] * siluf(zr[e]);
        gv[i] = g; s2 += g * g;
    }
    __shared__ float red[256];
    red[t] = s2; __syncthreads();
    for (int o = 128; o > 0; o >>= 1) {
        if (t < o) red[t] += red[t + o];
        __syncthreads();
    }
    float inv = rsqrtf(red[0] / (float)DI_ + 1e-5f);
    #pragma unroll
    for (int i = 0; i < 6; ++i) {
        int e = t + i * 256;
        G[(size_t)row * DI_ + e] = f2bf(gv[i] * inv * nw[e]);
    }
}

// ---------------- host-side orchestration ----------------
extern "C" void kernel_launch(void* const* d_in, const int* in_sizes, int n_in,
                              void* d_out, int out_size, void* d_ws, size_t ws_size,
                              hipStream_t stream) {
    const float* x       = (const float*)d_in[0];
    const float* ln_w    = (const float*)d_in[1];
    const float* ln_b    = (const float*)d_in[2];
    const float* Win     = (const float*)d_in[3];
    const float* conv_w  = (const float*)d_in[4];
    const float* conv_b  = (const float*)d_in[5];
    const float* dt_bias = (const float*)d_in[6];
    const float* A_log   = (const float*)d_in[7];
    const float* D_skip  = (const float*)d_in[8];
    const float* norm_w  = (const float*)d_in[9];
    const float* Wout    = (const float*)d_in[10];
    float* out = (float*)d_out;

    // carve workspace (256B aligned)
    char* p = (char*)d_ws;
    auto carve = [&](size_t bytes) {
        void* q = (void*)p;
        p += (bytes + 255) & ~(size_t)255;
        return q;
    };
    unsigned short* xnbf   = (unsigned short*)carve((size_t)MROWS_ * D_ * 2);
    unsigned short* Winbf  = (unsigned short*)carve((size_t)DIP_ * D_ * 2);
    unsigned short* Woutbf = (unsigned short*)carve((size_t)D_ * DI_ * 2);
    float*          Zbuf   = (float*)carve((size_t)MROWS_ * DIP_ * 4);
    float*          xBCc   = (float*)carve((size_t)MROWS_ * CD_ * 4);
    float*          dtb    = (float*)carve((size_t)MROWS_ * H_ * 4);
    float*          dAb    = (float*)carve((size_t)MROWS_ * H_ * 4);
    float*          Ybuf   = (float*)carve((size_t)MROWS_ * DI_ * 4);
    unsigned short* Gbuf   = (unsigned short*)carve((size_t)MROWS_ * DI_ * 2);
    (void)ws_size; (void)in_sizes; (void)n_in; (void)out_size;

    // weight casts
    {
        long long n1 = (long long)DIP_ * D_;
        k_cast_bf16<<<(unsigned)((n1 + 255) / 256), 256, 0, stream>>>(Win, Winbf, n1);
        long long n2 = (long long)D_ * DI_;
        k_cast_bf16<<<(unsigned)((n2 + 255) / 256), 256, 0, stream>>>(Wout, Woutbf, n2);
    }
    // LayerNorm + cast
    k_ln_cast<<<MROWS_, 256, 0, stream>>>(x, ln_w, ln_b, xnbf);

    // GEMM1: zxbcdt = xn @ Win^T   (M x DIP)
    {
        dim3 grid((DIP_ + BN - 1) / BN, MROWS_ / BM);
        k_gemm_bf16<<<grid, 256, 0, stream>>>(xnbf, Winbf, Zbuf, nullptr,
                                              MROWS_, D_, DIP_, 0);
    }
    // conv + silu
    {
        long long tot = (long long)MROWS_ * CD_;
        k_conv_silu<<<(unsigned)((tot + 255) / 256), 256, 0, stream>>>(Zbuf, conv_w, conv_b, xBCc);
    }
    // dt / dA
    {
        int tot = MROWS_ * H_;
        k_dt<<<(tot + 255) / 256, 256, 0, stream>>>(Zbuf, dt_bias, A_log, dtb, dAb);
    }
    // sequential scan: 192 persistent blocks
    k_scan<<<B_ * H_, 256, 0, stream>>>(xBCc, dtb, dAb, D_skip, Ybuf);
    // gate + rmsnorm + cast
    k_gate<<<MROWS_, 256, 0, stream>>>(Ybuf, Zbuf, norm_w, Gbuf);
    // GEMM2: out = g @ Wout^T + x  (M x D)
    {
        dim3 grid(D_ / BN, MROWS_ / BM);
        k_gemm_bf16<<<grid, 256, 0, stream>>>(Gbuf, Woutbf, out, x,
                                              MROWS_, DI_, D_, 1);
    }
}
